// DFAF_54099408060692
// MI455X (gfx1250) — compile-verified, matrix operand order checked by reading, not run on hardware
//
#include <hip/hip_runtime.h>
#include <hip/hip_bf16.h>
#include <math.h>

// ---------------------------------------------------------------------------
// Problem constants (from the reference)
// ---------------------------------------------------------------------------
#define BB   256
#define NO_  100
#define LL   30
#define VS_  2048
#define QS_  1280
#define OUT_ 512
#define HH   8
#define DH_  64
#define NB_  2

#define BNO (BB * NO_)   // 25600
#define BL  (BB * LL)    // 7680

// ---------------------------------------------------------------------------
// WMMA types (CDNA5 / gfx1250, wave32)
// ---------------------------------------------------------------------------
typedef __attribute__((ext_vector_type(16))) _Float16 v16h;
typedef __attribute__((ext_vector_type(8)))  _Float16 v8h;
typedef __attribute__((ext_vector_type(8)))  float    v8f;

// ---------------------------------------------------------------------------
// Generic GEMM: C[M,N] = act(A[M,K]) @ W[K,N] + bias, optional row-mask scale.
//   A = [A0 (K0 cols) | A1 (K1 cols)]  (A1 may be null; then K0 == K)
//   RELU template -> ReLU applied to A elements before the matmul.
// REQUIREMENT (satisfied by every call site): K0 % 32 == 0, so each K-tile is
// sourced entirely from A0 or entirely from A1 (pointer select once per tile).
//
// fp32 in/out, f16 WMMA with fp32 accumulation.
// Workgroup: 256 threads = 8 waves; block tile 128(M) x 64(N), K-step 32.
// Wave (wm in 0..3, wn in 0..1) owns rows wm*32..+31, cols wn*32..+31 as a
// 2x2 grid of 16x16 v_wmma_f32_16x16x32_f16 accumulators (A/B frag reuse).
// ---------------------------------------------------------------------------
#define GT_M 128
#define GT_N 64
#define GT_K 32
#define LPAD 40   // halves per LDS row: 80 B, multiple of 16 B -> b128 stores

template<int RELU>
__global__ __launch_bounds__(256)
void gemm_wmma_f16(const float* __restrict__ A0, int K0,
                   const float* __restrict__ A1, int K1,
                   const float* __restrict__ W,
                   const float* __restrict__ bias,
                   const float* __restrict__ rowmask,
                   float* __restrict__ C,
                   int M, int N, int K)
{
    __shared__ __align__(16) _Float16 As[GT_M][LPAD];
    __shared__ __align__(16) _Float16 Ws[GT_N][LPAD];   // W tile transposed: [n][k]

    const int tid  = threadIdx.x;
    const int lane = tid & 31;
    const int wave = tid >> 5;
    const int wm   = wave >> 1;          // 0..3 -> 32-row strip
    const int wn   = wave & 1;           // 0..1 -> 32-col strip
    const int blockN = blockIdx.x * GT_N;
    const int blockM = blockIdx.y * GT_M;

    v8f acc00 = {}, acc01 = {}, acc10 = {}, acc11 = {};

    // staging maps
    const int ar0   = tid >> 2;          // 0..63 : A rows ar0 and ar0+64
    const int acol0 = (tid & 3) << 3;    // 0,8,16,24 : 8 consecutive k (16B)
    const int wkn   = tid & 63;          // W: n within tile
    const int wk0   = (tid >> 6) << 3;   // W: k base 0,8,16,24

    // fragment constants (ISA 7.12.2, 16-bit A 16x32 layout)
    const int khi = (lane & 16) ? 8 : 0;
    const int lr  = lane & 15;

    for (int kt = 0; kt < K; kt += GT_K) {
        // ---- per-tile A-source select (uniform: K0 % GT_K == 0) ----
        const float* __restrict__ Ap;
        int Kp, koff;
        if (kt < K0) { Ap = A0; Kp = K0; koff = kt; }
        else         { Ap = A1; Kp = K1; koff = kt - K0; }

        // ---- stage A tile: 2 rows x 8 cols per thread, float4 loads ----
        {
#pragma unroll
            for (int rr = 0; rr < 2; ++rr) {
                const int row = ar0 + rr * 64;
                const float4 x = *(const float4*)&Ap[(size_t)(blockM + row) * Kp + koff + acol0];
                const float4 y = *(const float4*)&Ap[(size_t)(blockM + row) * Kp + koff + acol0 + 4];
                v8h hv;
                if (RELU) {
                    hv[0] = (_Float16)fmaxf(x.x, 0.0f); hv[1] = (_Float16)fmaxf(x.y, 0.0f);
                    hv[2] = (_Float16)fmaxf(x.z, 0.0f); hv[3] = (_Float16)fmaxf(x.w, 0.0f);
                    hv[4] = (_Float16)fmaxf(y.x, 0.0f); hv[5] = (_Float16)fmaxf(y.y, 0.0f);
                    hv[6] = (_Float16)fmaxf(y.z, 0.0f); hv[7] = (_Float16)fmaxf(y.w, 0.0f);
                } else {
                    hv[0] = (_Float16)x.x; hv[1] = (_Float16)x.y;
                    hv[2] = (_Float16)x.z; hv[3] = (_Float16)x.w;
                    hv[4] = (_Float16)y.x; hv[5] = (_Float16)y.y;
                    hv[6] = (_Float16)y.z; hv[7] = (_Float16)y.w;
                }
                *(v8h*)&As[row][acol0] = hv;    // one ds_store_b128
            }
        }
        // ---- stage W tile transposed: 8 independent b32 loads, one b128 store
        {
            float wv[8];
#pragma unroll
            for (int j = 0; j < 8; ++j)
                wv[j] = W[(size_t)(kt + wk0 + j) * N + blockN + wkn];
            v8h hw;
#pragma unroll
            for (int j = 0; j < 8; ++j) hw[j] = (_Float16)wv[j];
            *(v8h*)&Ws[wkn][wk0] = hw;
        }
        __syncthreads();

        // ---- prefetch next K-tile (gfx1250 global_prefetch_b8) ----
        if (kt + GT_K < K) {
            const int nkt = kt + GT_K;
            const float* __restrict__ Pp;
            int PK, pko;
            if (nkt < K0) { Pp = A0; PK = K0; pko = nkt; }
            else          { Pp = A1; PK = K1; pko = nkt - K0; }
            __builtin_prefetch(&Pp[(size_t)(blockM + ar0) * PK + pko + acol0], 0, 1);
            __builtin_prefetch(&W[(size_t)(nkt + wk0) * N + blockN + wkn], 0, 1);
        }

        // ---- gather fragments from LDS (ds_load_2addr_b32 pairs) ----
        v16h a0, a1, b0, b1;
        const _Float16* ap0 = &As[wm * 32 + lr][0];
        const _Float16* ap1 = &As[wm * 32 + 16 + lr][0];
        const _Float16* bp0 = &Ws[wn * 32 + lr][0];
        const _Float16* bp1 = &Ws[wn * 32 + 16 + lr][0];
#pragma unroll
        for (int p = 0; p < 8; ++p) {
            const int kb = ((p & 4) ? 16 : 0) + khi + ((p & 3) << 1);
            a0[2 * p] = ap0[kb]; a0[2 * p + 1] = ap0[kb + 1];
            a1[2 * p] = ap1[kb]; a1[2 * p + 1] = ap1[kb + 1];
            b0[2 * p] = bp0[kb]; b0[2 * p + 1] = bp0[kb + 1];
            b1[2 * p] = bp1[kb]; b1[2 * p + 1] = bp1[kb + 1];
        }

        acc00 = __builtin_amdgcn_wmma_f32_16x16x32_f16(false, a0, false, b0,
                                                       (short)0, acc00, false, false);
        acc01 = __builtin_amdgcn_wmma_f32_16x16x32_f16(false, a0, false, b1,
                                                       (short)0, acc01, false, false);
        acc10 = __builtin_amdgcn_wmma_f32_16x16x32_f16(false, a1, false, b0,
                                                       (short)0, acc10, false, false);
        acc11 = __builtin_amdgcn_wmma_f32_16x16x32_f16(false, a1, false, b1,
                                                       (short)0, acc11, false, false);
        __syncthreads();
    }

    // ---- epilogue: C/D layout (VGPR r: lanes 0-15 -> M=r, lanes 16-31 -> M=r+8)
    const int mof  = (lane & 16) ? 8 : 0;
    const int nof  = lane & 15;
    const int col0 = blockN + wn * 32 + nof;
    const int col1 = col0 + 16;
    const float bb0 = bias ? bias[col0] : 0.0f;
    const float bb1 = bias ? bias[col1] : 0.0f;
    const int rbase0 = blockM + wm * 32 + mof;
    const int rbase1 = rbase0 + 16;
#pragma unroll
    for (int r = 0; r < 8; ++r) {
        const int row0 = rbase0 + r;
        const int row1 = rbase1 + r;
        const float m0 = rowmask ? rowmask[row0] : 1.0f;
        const float m1 = rowmask ? rowmask[row1] : 1.0f;
        C[(size_t)row0 * N + col0] = (acc00[r] + bb0) * m0;
        C[(size_t)row0 * N + col1] = (acc01[r] + bb1) * m0;
        C[(size_t)row1 * N + col0] = (acc10[r] + bb0) * m1;
        C[(size_t)row1 * N + col1] = (acc11[r] + bb1) * m1;
    }
}

// ---------------------------------------------------------------------------
// Multi-head attention, one workgroup per (b, h). K/V/mask/scores in LDS.
// Wave-per-query, lanes own 2 channels of DH=64; matches reference exactly:
// masked_fill(-1e9) BEFORE the /sqrt(64) scaling, fp32 softmax.
// ---------------------------------------------------------------------------
#define MAXK 112   // >= max(NO_, LL), padded

__global__ __launch_bounds__(256)
void mha_attn(const float* __restrict__ Qb_, int qs,
              const float* __restrict__ Kb_, int ks_,
              const float* __restrict__ Vb_, int vs_,
              const float* __restrict__ keymask,
              float* __restrict__ Out,
              int Nq, int Nk)
{
    __shared__ float Ks[MAXK][DH_];
    __shared__ float Vs[MAXK][DH_];
    __shared__ float Sc[8][MAXK];
    __shared__ float Msk[MAXK];

    const int tid  = threadIdx.x;
    const int lane = tid & 31;
    const int wave = tid >> 5;
    const int b = blockIdx.x >> 3;    // H == 8
    const int h = blockIdx.x & 7;

    // cooperative load of K, V (this head) and key mask into LDS
    for (int i = tid; i < Nk * DH_; i += 256) {
        const int m = i >> 6;
        const int d = i & 63;
        Ks[m][d] = Kb_[(size_t)(b * Nk + m) * ks_ + h * DH_ + d];
        Vs[m][d] = Vb_[(size_t)(b * Nk + m) * vs_ + h * DH_ + d];
    }
    for (int m = tid; m < Nk; m += 256) Msk[m] = keymask[b * Nk + m];
    __syncthreads();

    const int d0 = 2 * lane;
    const int d1 = d0 + 1;

    for (int n = wave; n < Nq; n += 8) {
        const float* qp = Qb_ + (size_t)(b * Nq + n) * qs + h * DH_;
        const float q0 = qp[d0];
        const float q1 = qp[d1];

        // scores: wave-cooperative 64-dot per key
        for (int m = 0; m < Nk; ++m) {
            float part = q0 * Ks[m][d0] + q1 * Ks[m][d1];
#pragma unroll
            for (int off = 16; off; off >>= 1) part += __shfl_xor(part, off);
            float sc = (Msk[m] == 0.0f) ? -1e9f : part;   // mask BEFORE scaling
            sc *= 0.125f;                                 // 1/sqrt(64)
            if (lane == 0) Sc[wave][m] = sc;
        }

        // softmax over keys
        float mx = -3.4e38f;
        for (int m = lane; m < Nk; m += 32) mx = fmaxf(mx, Sc[wave][m]);
#pragma unroll
        for (int off = 16; off; off >>= 1) mx = fmaxf(mx, __shfl_xor(mx, off));

        float ssum = 0.0f;
        for (int m = lane; m < Nk; m += 32) {
            const float p = __expf(Sc[wave][m] - mx);
            Sc[wave][m] = p;
            ssum += p;
        }
#pragma unroll
        for (int off = 16; off; off >>= 1) ssum += __shfl_xor(ssum, off);
        const float inv = 1.0f / ssum;

        // context: lanes own 2 output channels
        float o0 = 0.0f, o1 = 0.0f;
        for (int m = 0; m < Nk; ++m) {
            const float p = Sc[wave][m] * inv;
            o0 += p * Vs[m][d0];
            o1 += p * Vs[m][d1];
        }
        float* op = Out + (size_t)(b * Nq + n) * OUT_ + h * DH_;
        op[d0] = o0;
        op[d1] = o1;
    }
}

// ---------------------------------------------------------------------------
// Elementwise helpers
// ---------------------------------------------------------------------------
__global__ void masked_mean_kernel(const float* __restrict__ X,
                                   const float* __restrict__ mask,
                                   float* __restrict__ out, int Nn)
{
    const int b = blockIdx.x;
    const int c = threadIdx.x;        // blockDim.x == OUT_
    const float* x  = X + (size_t)b * Nn * OUT_ + c;
    const float* mk = mask + (size_t)b * Nn;
    float s = 0.0f, d = 0.0f;
    for (int n = 0; n < Nn; ++n) {
        s += x[(size_t)n * OUT_] * mk[n];
        d += mk[n];
    }
    out[(size_t)b * OUT_ + c] = s / d;
}

__global__ void sigmoid_ip_kernel(float* __restrict__ x, int n)
{
    const int i = blockIdx.x * blockDim.x + threadIdx.x;
    if (i < n) x[i] = 1.0f / (1.0f + __expf(-x[i]));
}

// Scale k- and q- parts of T[row, 0:1024] by (1 + gate[b, c]); v-part untouched.
__global__ void gate_scale_kernel(float* __restrict__ T,
                                  const float* __restrict__ gate,
                                  int rowsPerB, int totalRows)
{
    const int i = blockIdx.x * blockDim.x + threadIdx.x;
    if (i >= totalRows * OUT_) return;
    const int row = i >> 9;           // /512
    const int c   = i & 511;
    const int b   = row / rowsPerB;
    const float g = 1.0f + gate[(size_t)b * OUT_ + c];
    float* p = T + (size_t)row * (3 * OUT_);
    p[c]        *= g;                 // k part
    p[OUT_ + c] *= g;                 // q part
}

__global__ void add_ip_kernel(float* __restrict__ y, const float* __restrict__ x, int n)
{
    const int i = blockIdx.x * blockDim.x + threadIdx.x;
    if (i < n) y[i] += x[i];
}

// F.normalize(x, p=2, dim=1): normalize over the sequence dim, per (b, channel)
__global__ void l2norm_dim1_kernel(const float* __restrict__ X,
                                   float* __restrict__ Y, int Nn)
{
    const int idx = blockIdx.x * blockDim.x + threadIdx.x; // b*512 + c
    if (idx >= BB * OUT_) return;
    const int b = idx >> 9;
    const int c = idx & 511;
    const float* x = X + (size_t)b * Nn * OUT_ + c;
    float s = 0.0f;
    for (int n = 0; n < Nn; ++n) {
        const float v = x[(size_t)n * OUT_];
        s += v * v;
    }
    const float inv = 1.0f / fmaxf(sqrtf(s), 1e-12f);
    float* y = Y + (size_t)b * Nn * OUT_ + c;
    for (int n = 0; n < Nn; ++n) y[(size_t)n * OUT_] = x[(size_t)n * OUT_] * inv;
}

// ---------------------------------------------------------------------------
// Host orchestration
// ---------------------------------------------------------------------------
static inline void run_gemm(hipStream_t s, float* C,
                            const float* A0, int K0, const float* A1, int K1,
                            const float* W, const float* bias, const float* rmask,
                            int M, int N, int K, int relu)
{
    dim3 g(N / GT_N, M / GT_M);
    if (relu)
        gemm_wmma_f16<1><<<g, 256, 0, s>>>(A0, K0, A1, K1, W, bias, rmask, C, M, N, K);
    else
        gemm_wmma_f16<0><<<g, 256, 0, s>>>(A0, K0, A1, K1, W, bias, rmask, C, M, N, K);
}

extern "C" void kernel_launch(void* const* d_in, const int* in_sizes, int n_in,
                              void* d_out, int out_size, void* d_ws, size_t ws_size,
                              hipStream_t stream)
{
    (void)in_sizes; (void)n_in; (void)out_size; (void)ws_size;

    const float* v_in   = (const float*)d_in[0];   // [B,NO,VS]
    const float* q_in   = (const float*)d_in[1];   // [B,L,QS]
    const float* q_mask = (const float*)d_in[2];   // [B,L]
    const float* v_mask = (const float*)d_in[3];   // [B,NO]
    // params in dict order
    const float* vlin_w   = (const float*)d_in[4];
    const float* vlin_b   = (const float*)d_in[5];
    const float* qlin_w   = (const float*)d_in[6];
    const float* qlin_b   = (const float*)d_in[7];
    const float* i_vlin_w = (const float*)d_in[8];
    const float* i_vlin_b = (const float*)d_in[9];
    const float* i_qlin_w = (const float*)d_in[10];
    const float* i_qlin_b = (const float*)d_in[11];
    const float* i_vout_w = (const float*)d_in[12];
    const float* i_vout_b = (const float*)d_in[13];
    const float* i_qout_w = (const float*)d_in[14];
    const float* i_qout_b = (const float*)d_in[15];
    const float* d_v4q_w  = (const float*)d_in[16];
    const float* d_v4q_b  = (const float*)d_in[17];
    const float* d_q4v_w  = (const float*)d_in[18];
    const float* d_q4v_b  = (const float*)d_in[19];
    const float* d_vlin_w = (const float*)d_in[20];
    const float* d_vlin_b = (const float*)d_in[21];
    const float* d_qlin_w = (const float*)d_in[22];
    const float* d_qlin_b = (const float*)d_in[23];
    const float* d_vout_w = (const float*)d_in[24];
    const float* d_vout_b = (const float*)d_in[25];
    const float* d_qout_w = (const float*)d_in[26];
    const float* d_qout_b = (const float*)d_in[27];

    float* out = (float*)d_out;

    // workspace layout (fp32)
    float* ws = (float*)d_ws;
    size_t o = 0;
    float* Va    = ws + o; o += (size_t)BNO * OUT_;
    float* Vb    = ws + o; o += (size_t)BNO * OUT_;
    float* Qa    = ws + o; o += (size_t)BL  * OUT_;
    float* Qb    = ws + o; o += (size_t)BL  * OUT_;
    float* vt    = ws + o; o += (size_t)BNO * 3 * OUT_;
    float* qt    = ws + o; o += (size_t)BL  * 3 * OUT_;
    float* vupd  = ws + o; o += (size_t)BNO * OUT_;
    float* qupd  = ws + o; o += (size_t)BL  * OUT_;
    float* vmean = ws + o; o += (size_t)BB * OUT_;
    float* qmean = ws + o; o += (size_t)BB * OUT_;
    float* gv4q  = ws + o; o += (size_t)BB * OUT_;
    float* gq4v  = ws + o; o += (size_t)BB * OUT_;

    // ---- input projections ----
    run_gemm(stream, Va, v_in, VS_, nullptr, 0, vlin_w, vlin_b, nullptr,
             BNO, OUT_, VS_, 0);
    run_gemm(stream, Qa, q_in, QS_, nullptr, 0, qlin_w, qlin_b, nullptr,
             BL, OUT_, QS_, 0);

    for (int blk = 0; blk < NB_; ++blk) {
        // ================= inter =================
        run_gemm(stream, vt, Va, OUT_, nullptr, 0, i_vlin_w, i_vlin_b, v_mask,
                 BNO, 3 * OUT_, OUT_, 1);
        run_gemm(stream, qt, Qa, OUT_, nullptr, 0, i_qlin_w, i_qlin_b, q_mask,
                 BL, 3 * OUT_, OUT_, 1);
        // v attends to q: Q=v_q, K=q_k, V=q_v, mask=q_mask
        mha_attn<<<BB * HH, 256, 0, stream>>>(vt + OUT_, 3 * OUT_,
                                              qt,        3 * OUT_,
                                              qt + 2 * OUT_, 3 * OUT_,
                                              q_mask, vupd, NO_, LL);
        // q attends to v
        mha_attn<<<BB * HH, 256, 0, stream>>>(qt + OUT_, 3 * OUT_,
                                              vt,        3 * OUT_,
                                              vt + 2 * OUT_, 3 * OUT_,
                                              v_mask, qupd, LL, NO_);
        // new_v = concat([v, v_update]) @ i_vout + b  (dual-A GEMM)
        run_gemm(stream, Vb, Va, OUT_, vupd, OUT_, i_vout_w, i_vout_b, nullptr,
                 BNO, OUT_, 2 * OUT_, 0);
        run_gemm(stream, Qb, Qa, OUT_, qupd, OUT_, i_qout_w, i_qout_b, nullptr,
                 BL, OUT_, 2 * OUT_, 0);

        // ================= intra =================
        masked_mean_kernel<<<BB, OUT_, 0, stream>>>(Vb, v_mask, vmean, NO_);
        masked_mean_kernel<<<BB, OUT_, 0, stream>>>(Qb, q_mask, qmean, LL);
        run_gemm(stream, gv4q, vmean, OUT_, nullptr, 0, d_v4q_w, d_v4q_b, nullptr,
                 BB, OUT_, OUT_, 1);
        run_gemm(stream, gq4v, qmean, OUT_, nullptr, 0, d_q4v_w, d_q4v_b, nullptr,
                 BB, OUT_, OUT_, 1);
        {
            const int n = BB * OUT_;
            sigmoid_ip_kernel<<<(n + 255) / 256, 256, 0, stream>>>(gv4q, n);
            sigmoid_ip_kernel<<<(n + 255) / 256, 256, 0, stream>>>(gq4v, n);
        }
        run_gemm(stream, vt, Vb, OUT_, nullptr, 0, d_vlin_w, d_vlin_b, v_mask,
                 BNO, 3 * OUT_, OUT_, 1);
        run_gemm(stream, qt, Qb, OUT_, nullptr, 0, d_qlin_w, d_qlin_b, q_mask,
                 BL, 3 * OUT_, OUT_, 1);
        // gate k/q parts: v side gated by q4v, q side gated by v4q
        {
            const int nv = BNO * OUT_;
            gate_scale_kernel<<<(nv + 255) / 256, 256, 0, stream>>>(vt, gq4v, NO_, BNO);
            const int nq = BL * OUT_;
            gate_scale_kernel<<<(nq + 255) / 256, 256, 0, stream>>>(qt, gv4q, LL, BL);
        }
        // self-attention within each stream
        mha_attn<<<BB * HH, 256, 0, stream>>>(vt + OUT_, 3 * OUT_,
                                              vt,        3 * OUT_,
                                              vt + 2 * OUT_, 3 * OUT_,
                                              v_mask, vupd, NO_, NO_);
        mha_attn<<<BB * HH, 256, 0, stream>>>(qt + OUT_, 3 * OUT_,
                                              qt,        3 * OUT_,
                                              qt + 2 * OUT_, 3 * OUT_,
                                              q_mask, qupd, LL, LL);
        // residual: v_upd += v ; then out-projection
        {
            const int nv = BNO * OUT_;
            add_ip_kernel<<<(nv + 255) / 256, 256, 0, stream>>>(vupd, Vb, nv);
            const int nq = BL * OUT_;
            add_ip_kernel<<<(nq + 255) / 256, 256, 0, stream>>>(qupd, Qb, nq);
        }
        run_gemm(stream, Vb, vupd, OUT_, nullptr, 0, d_vout_w, d_vout_b, nullptr,
                 BNO, OUT_, OUT_, 0);
        run_gemm(stream, Qb, qupd, OUT_, nullptr, 0, d_qout_w, d_qout_b, nullptr,
                 BL, OUT_, OUT_, 0);

        // ---- L2 norm over dim 1 ----
        float* dstV = (blk == NB_ - 1) ? out : Va;
        float* dstQ = (blk == NB_ - 1) ? (out + (size_t)BNO * OUT_) : Qa;
        {
            const int n = BB * OUT_;
            l2norm_dim1_kernel<<<(n + 255) / 256, 256, 0, stream>>>(Vb, dstV, NO_);
            l2norm_dim1_kernel<<<(n + 255) / 256, 256, 0, stream>>>(Qb, dstQ, LL);
        }
    }
}
